// MultiHeadAttention_16698832847024
// MI455X (gfx1250) — matrix operand order repaired and checked by reading, not compiled, not run on hardware
//
#include <hip/hip_runtime.h>
#include <hip/hip_bf16.h>
#include <math.h>

// ---------------------------------------------------------------------------
// MHA forward for B=4, L=2048, D=768, H=12, HD=64 on gfx1250 (MI455X).
// f16 data / f32 accum via v_wmma_f32_16x16x32_f16; flash-attention online
// softmax in f32 registers (never materializes the LxL matrix).
// ---------------------------------------------------------------------------

typedef __attribute__((ext_vector_type(16))) _Float16 v16h;
typedef __attribute__((ext_vector_type(8)))  float    v8f;

#define B_   4
#define L_   2048
#define D_   768
#define H_   12
#define HD_  64
#define BH_  (B_ * H_)      // 48
#define M_   (B_ * L_)      // 8192
#define N1_  (3 * D_)       // 2304

union FragU {
    unsigned int u[8];
    uint4        q[2];
    v16h         h;
};

// ---------------------------------------------------------------------------
// WMMA fragment loaders (16-bit A 16x32 and B 32x16 layouts per ISA 7.12.2).
// A: lane = {half=lane/16, m=lane%16}; dword d: d<4 -> k pair (2d + 8*half),
//    d>=4 -> (16 + 2(d-4) + 8*half).
// B: lane = {khalf=lane/16, n=lane%16}; k 8*khalf..+7 then 16+8*khalf..+7.
// ---------------------------------------------------------------------------
__device__ inline v16h load_a_frag(const _Float16* __restrict__ rowsBase,
                                   int rowStrideHalves, int lane) {
    const int m    = lane & 15;
    const int half = lane >> 4;
    const _Float16* base = rowsBase + (size_t)m * rowStrideHalves + half * 8;
    FragU r;
    const unsigned int* p0 = (const unsigned int*)base;        // k rel 0..7
    const unsigned int* p1 = (const unsigned int*)(base + 16); // k rel 16..23
    r.u[0] = p0[0]; r.u[1] = p0[1]; r.u[2] = p0[2]; r.u[3] = p0[3];
    r.u[4] = p1[0]; r.u[5] = p1[1]; r.u[6] = p1[2]; r.u[7] = p1[3];
    return r.h;
}

// B operand: element (k, n) at colBase[n * colStrideHalves + k].
__device__ inline v16h load_b_frag(const _Float16* __restrict__ colBase,
                                   int colStrideHalves, int lane) {
    const int n     = lane & 15;
    const int khalf = lane >> 4;
    const _Float16* base = colBase + (size_t)n * colStrideHalves + khalf * 8;
    FragU r;
    r.q[0] = *(const uint4*)base;
    r.q[1] = *(const uint4*)(base + 16);
    return r.h;
}

// ---------------------------------------------------------------------------
// Conversion kernels
// ---------------------------------------------------------------------------
__global__ void f32_to_f16_kernel(const float* __restrict__ in,
                                  _Float16* __restrict__ out, int n) {
    int i = blockIdx.x * blockDim.x + threadIdx.x;
    if (i < n) out[i] = (_Float16)in[i];
}

// in: [K, N] f32 row-major  ->  out: [N, K] f16 row-major
__global__ void transpose_f32_to_f16_kernel(const float* __restrict__ in,
                                            _Float16* __restrict__ out,
                                            int K, int N) {
    int i = blockIdx.x * blockDim.x + threadIdx.x;
    if (i < K * N) {
        int k = i / N, n = i % N;
        out[(size_t)n * K + k] = (_Float16)in[i];
    }
}

// ---------------------------------------------------------------------------
// Tiled GEMM: C[M,N] = A[M,K](f16 row-major) * Bt[N,K](f16 row-major)^T
//             + bias[N](f32).  N, K, output type are compile-time.
// Block 256 thr = 8 waves; tile 128x128; K-step 32; double-buffered LDS.
// Wave grid 2(M) x 4(N); wave tile 64x32 => 4x2 WMMA accumulators.
// ---------------------------------------------------------------------------
#define LDT 40  // LDS row stride in halves (80B rows, 16B-aligned)

template <int N, int K, bool F32OUT>
__global__ __launch_bounds__(256)
void gemm_f16_kernel(const _Float16* __restrict__ A,
                     const _Float16* __restrict__ Bt,
                     const float* __restrict__ bias,
                     void* __restrict__ outp) {
    __shared__ _Float16 As[2][128 * LDT];
    __shared__ _Float16 Bs[2][128 * LDT];

    const int tid  = threadIdx.x;
    const int lane = tid & 31;
    const int w    = tid >> 5;
    const int wm   = w & 1;        // M offset wm*64
    const int wn   = w >> 1;       // N offset wn*32
    const int m0   = blockIdx.y * 128;
    const int n0   = blockIdx.x * 128;

    v8f acc[4][2];
    {
        v8f z = {};
        for (int i = 0; i < 4; i++)
            for (int j = 0; j < 2; j++) acc[i][j] = z;
    }

    const int r  = tid >> 1;  // 0..127 tile row
    const int hh = tid & 1;   // 16-half chunk of the 32-wide K slice

    const _Float16* gArow = A  + (size_t)(m0 + r) * K + hh * 16;
    const _Float16* gBrow = Bt + (size_t)(n0 + r) * K + hh * 16;

    auto load_tile = [&](int kt, int buf) {
        const uint4* ga = (const uint4*)(gArow + kt * 32);
        uint4* da = (uint4*)&As[buf][r * LDT + hh * 16];
        da[0] = ga[0]; da[1] = ga[1];
        const uint4* gb = (const uint4*)(gBrow + kt * 32);
        uint4* db = (uint4*)&Bs[buf][r * LDT + hh * 16];
        db[0] = gb[0]; db[1] = gb[1];
    };

    constexpr int NKT = K / 32;  // 24
    load_tile(0, 0);
    __syncthreads();

    for (int kt = 0; kt < NKT; kt++) {
        const int cur = kt & 1;
        if (kt + 1 < NKT) load_tile(kt + 1, cur ^ 1);
        if (kt + 2 < NKT) {  // warm GL2 behind the double buffer
            __builtin_prefetch(gArow + (kt + 2) * 32, 0, 1);
            __builtin_prefetch(gBrow + (kt + 2) * 32, 0, 1);
        }

        v16h bfrag[2];
        #pragma unroll
        for (int nf = 0; nf < 2; nf++)
            bfrag[nf] =
                load_b_frag(&Bs[cur][(wn * 32 + nf * 16) * LDT], LDT, lane);

        #pragma unroll
        for (int mf = 0; mf < 4; mf++) {
            v16h afrag =
                load_a_frag(&As[cur][(wm * 64 + mf * 16) * LDT], LDT, lane);
            #pragma unroll
            for (int nf = 0; nf < 2; nf++)
                acc[mf][nf] = __builtin_amdgcn_wmma_f32_16x16x32_f16(
                    false, afrag, false, bfrag[nf], (short)0, acc[mf][nf],
                    false, false);
        }
        __syncthreads();
    }

    // Epilogue. C layout: slot s, lane -> row = s + 8*(lane/16), col = lane%16.
    const int nl   = lane & 15;
    const int half = lane >> 4;
    float bv[2];
    #pragma unroll
    for (int nf = 0; nf < 2; nf++)
        bv[nf] = bias[n0 + wn * 32 + nf * 16 + nl];

    #pragma unroll
    for (int mf = 0; mf < 4; mf++) {
        const size_t rbase = (size_t)(m0 + wm * 64 + mf * 16 + 8 * half) * N;
        #pragma unroll
        for (int nf = 0; nf < 2; nf++) {
            const int col = n0 + wn * 32 + nf * 16 + nl;
            if constexpr (F32OUT) {
                float* p = (float*)outp + rbase + col;
                #pragma unroll
                for (int s = 0; s < 8; s++)
                    p[(size_t)s * N] = acc[mf][nf][s] + bv[nf];
            } else {
                _Float16* p = (_Float16*)outp + rbase + col;
                #pragma unroll
                for (int s = 0; s < 8; s++)
                    p[(size_t)s * N] = (_Float16)(acc[mf][nf][s] + bv[nf]);
            }
        }
    }
}

// ---------------------------------------------------------------------------
// RoPE + head split.  qkv: [B*L, 3*D] f16.  Outputs:
//   Qr, Kr : [BH, L, 64] f16 (rotated)
//   Vt     : [BH, 64, L] f16 (transposed so PV B-fragments are contiguous)
// ---------------------------------------------------------------------------
__global__ void rope_split_kernel(const _Float16* __restrict__ qkv,
                                  _Float16* __restrict__ Qr,
                                  _Float16* __restrict__ Kr,
                                  _Float16* __restrict__ Vt) {
    int idx = blockIdx.x * blockDim.x + threadIdx.x;  // over BH_*L_
    if (idx >= BH_ * L_) return;
    const int l  = idx % L_;
    const int bh = idx / L_;
    const int b  = bh / H_;
    const int h  = bh % H_;

    const _Float16* row = qkv + (size_t)(b * L_ + l) * N1_;
    const _Float16* qp  = row + h * HD_;
    const _Float16* kp  = row + D_ + h * HD_;
    const _Float16* vp  = row + 2 * D_ + h * HD_;
    _Float16* qo = Qr + (size_t)idx * HD_;
    _Float16* ko = Kr + (size_t)idx * HD_;

    float q[HD_], k[HD_];
    #pragma unroll 8
    for (int d = 0; d < HD_; d++) { q[d] = (float)qp[d]; k[d] = (float)kp[d]; }

    const float fl = (float)l;
    // inv_freq[i] = 10000^(-i/32) = 2^(-i * log2(10000)/32)
    const float c0 = 0.4152410118609203f;  // log2(10000)/32
    for (int i = 0; i < HD_ / 2; i++) {
        const float invf = exp2f(-c0 * (float)i);
        float s, c;
        sincosf(fl * invf, &s, &c);
        qo[i]      = (_Float16)(q[i] * c      - q[i + 32] * s);
        qo[i + 32] = (_Float16)(q[i + 32] * c + q[i] * s);
        ko[i]      = (_Float16)(k[i] * c      - k[i + 32] * s);
        ko[i + 32] = (_Float16)(k[i + 32] * c + k[i] * s);
    }

    _Float16* vo = Vt + (size_t)bh * HD_ * L_;
    #pragma unroll 8
    for (int d = 0; d < HD_; d++) vo[(size_t)d * L_ + l] = vp[d];
}

// ---------------------------------------------------------------------------
// Flash attention.  Grid: (L/128, BH).  Block: 256 threads = 8 waves.
// Per wave: 16-row Q block, 32-key tiles, causal trip count, online softmax,
// P re-swizzled via wave-private LDS into WMMA A-fragment layout.
// ---------------------------------------------------------------------------
__global__ __launch_bounds__(256)
void flash_attn_kernel(const _Float16* __restrict__ Qr,
                       const _Float16* __restrict__ Kr,
                       const _Float16* __restrict__ Vt,
                       const unsigned char* __restrict__ pad,
                       _Float16* __restrict__ AO) {
    __shared__ _Float16 Ptile[8][16 * LDT];  // per-wave 16x32 P tile

    const int tid  = threadIdx.x;
    const int lane = tid & 31;
    const int w    = tid >> 5;
    const int bh   = blockIdx.y;
    const int b    = bh / H_;
    const int h    = bh % H_;
    const int q0   = blockIdx.x * 128 + w * 16;
    const int nl   = lane & 15;
    const int half = lane >> 4;
    const float scale = 0.125f;  // 1/sqrt(64)

    // Q A-fragments (d 0..31 and 32..63), loaded once from global.
    const _Float16* qbase = Qr + ((size_t)bh * L_ + q0 + nl) * HD_ + half * 8;
    v16h aq[2];
    #pragma unroll
    for (int f = 0; f < 2; f++) {
        FragU r;
        const unsigned int* p0 = (const unsigned int*)(qbase + f * 32);
        const unsigned int* p1 = (const unsigned int*)(qbase + f * 32 + 16);
        r.u[0] = p0[0]; r.u[1] = p0[1]; r.u[2] = p0[2]; r.u[3] = p0[3];
        r.u[4] = p1[0]; r.u[5] = p1[1]; r.u[6] = p1[2]; r.u[7] = p1[3];
        aq[f] = r.h;
    }

    float mrow[8], lrow[8];
    v8f o[4];
    {
        v8f z = {};
        for (int s = 0; s < 8; s++) { mrow[s] = -3.0e38f; lrow[s] = 0.0f; }
        for (int dt = 0; dt < 4; dt++) o[dt] = z;
    }

    _Float16* pt = &Ptile[w][0];
    const int nkt = (q0 + 15) / 32 + 1;  // causal: key tiles up to max row

    for (int kt = 0; kt < nkt; kt++) {
        const int keyb = kt * 32;
        v8f s0 = {}, s1 = {};

        // S = Q K^T : B element (k=d, n=key) = Kr[key][d]
        #pragma unroll
        for (int dhf = 0; dhf < 2; dhf++) {
            const _Float16* kb0 =
                Kr + ((size_t)bh * L_ + keyb) * HD_ + dhf * 32;
            v16h b0 = load_b_frag(kb0, HD_, lane);
            v16h b1 = load_b_frag(kb0 + 16 * HD_, HD_, lane);
            s0 = __builtin_amdgcn_wmma_f32_16x16x32_f16(
                false, aq[dhf], false, b0, (short)0, s0, false, false);
            s1 = __builtin_amdgcn_wmma_f32_16x16x32_f16(
                false, aq[dhf], false, b1, (short)0, s1, false, false);
        }

        const int col0 = keyb + nl, col1 = col0 + 16;
        const bool pad0 = pad[b * L_ + col0] != 0;
        const bool pad1 = pad[b * L_ + col1] != 0;

        // Online softmax: slot s holds row q0+s+8*half; its 16 columns live
        // in one 16-lane group -> xor-shuffle reductions with masks < 16.
        #pragma unroll
        for (int s = 0; s < 8; s++) {
            const int row = q0 + s + 8 * half;
            float x0 = s0[s] * scale;
            float x1 = s1[s] * scale;
            if (pad0 || col0 > row) x0 = -3.0e38f;
            if (pad1 || col1 > row) x1 = -3.0e38f;

            float t = fmaxf(x0, x1);
            #pragma unroll
            for (int off = 1; off < 16; off <<= 1)
                t = fmaxf(t, __shfl_xor(t, off, 32));
            const float mn = fmaxf(mrow[s], t);
            const float p0 = __expf(x0 - mn);
            const float p1 = __expf(x1 - mn);
            float rs = p0 + p1;
            #pragma unroll
            for (int off = 1; off < 16; off <<= 1)
                rs += __shfl_xor(rs, off, 32);
            const float alpha = __expf(mrow[s] - mn);
            lrow[s] = lrow[s] * alpha + rs;
            mrow[s] = mn;
            #pragma unroll
            for (int dt = 0; dt < 4; dt++) o[dt][s] *= alpha;

            pt[(s + 8 * half) * LDT + nl]      = (_Float16)p0;
            pt[(s + 8 * half) * LDT + nl + 16] = (_Float16)p1;
        }
        // Same-wave LDS ops are in order; compiler inserts s_wait_dscnt.
        v16h pa = load_a_frag(pt, LDT, lane);

        // O += P V : B element (k=key, n=d) = Vt[d][key]
        #pragma unroll
        for (int dt = 0; dt < 4; dt++) {
            const _Float16* vb =
                Vt + ((size_t)bh * HD_ + dt * 16) * L_ + keyb;
            v16h bv = load_b_frag(vb, L_, lane);
            o[dt] = __builtin_amdgcn_wmma_f32_16x16x32_f16(
                false, pa, false, bv, (short)0, o[dt], false, false);
        }
    }

    // Normalize and store as f16 [B, L, D] with heads re-interleaved.
    #pragma unroll
    for (int s = 0; s < 8; s++) {
        const int row = q0 + s + 8 * half;
        const float inv = 1.0f / lrow[s];
        #pragma unroll
        for (int dt = 0; dt < 4; dt++) {
            AO[(size_t)(b * L_ + row) * D_ + h * HD_ + dt * 16 + nl] =
                (_Float16)(o[dt][s] * inv);
        }
    }
}

// ---------------------------------------------------------------------------
// Host-side launch
// ---------------------------------------------------------------------------
extern "C" void kernel_launch(void* const* d_in, const int* in_sizes, int n_in,
                              void* d_out, int out_size, void* d_ws,
                              size_t ws_size, hipStream_t stream) {
    (void)in_sizes; (void)n_in; (void)out_size; (void)ws_size;

    const float*         X    = (const float*)d_in[0];
    const unsigned char* pad  = (const unsigned char*)d_in[1];  // bool mask
    const float*         Wqkv = (const float*)d_in[2];
    const float*         bqkv = (const float*)d_in[3];
    const float*         Wout = (const float*)d_in[4];
    const float*         bout = (const float*)d_in[5];
    float*               out  = (float*)d_out;

    char*  ws  = (char*)d_ws;
    size_t off = 0;
    auto alloc = [&](size_t bytes) -> void* {
        void* p = ws + off;
        off += (bytes + 255) & ~(size_t)255;
        return p;
    };

    _Float16* Xh    = (_Float16*)alloc((size_t)M_ * D_ * 2);
    _Float16* WqkvT = (_Float16*)alloc((size_t)N1_ * D_ * 2);
    _Float16* WoutT = (_Float16*)alloc((size_t)D_ * D_ * 2);
    _Float16* QKVh  = (_Float16*)alloc((size_t)M_ * N1_ * 2);
    _Float16* Qr    = (_Float16*)alloc((size_t)BH_ * L_ * HD_ * 2);
    _Float16* Kr    = (_Float16*)alloc((size_t)BH_ * L_ * HD_ * 2);
    _Float16* Vt    = (_Float16*)alloc((size_t)BH_ * L_ * HD_ * 2);
    _Float16* AOh   = (_Float16*)alloc((size_t)M_ * D_ * 2);

    int n;
    n = M_ * D_;
    f32_to_f16_kernel<<<(n + 255) / 256, 256, 0, stream>>>(X, Xh, n);
    n = D_ * N1_;
    transpose_f32_to_f16_kernel<<<(n + 255) / 256, 256, 0, stream>>>(
        Wqkv, WqkvT, D_, N1_);
    n = D_ * D_;
    transpose_f32_to_f16_kernel<<<(n + 255) / 256, 256, 0, stream>>>(
        Wout, WoutT, D_, D_);

    // QKV projection: [8192,768] x [768,2304] -> f16 [8192,2304]
    dim3 g1(N1_ / 128, M_ / 128);
    gemm_f16_kernel<N1_, D_, false>
        <<<g1, 256, 0, stream>>>(Xh, WqkvT, bqkv, QKVh);

    n = BH_ * L_;
    rope_split_kernel<<<(n + 255) / 256, 256, 0, stream>>>(QKVh, Qr, Kr, Vt);

    dim3 gf(L_ / 128, BH_);
    flash_attn_kernel<<<gf, 256, 0, stream>>>(Qr, Kr, Vt, pad, AOh);

    // Output projection: [8192,768] x [768,768] -> f32 d_out (+bias)
    dim3 g2(D_ / 128, M_ / 128);
    gemm_f16_kernel<D_, D_, true>
        <<<g2, 256, 0, stream>>>(AOh, WoutT, bout, out);
}
